// Spatial2DNATBlock_58987080843391
// MI455X (gfx1250) — matrix operand order repaired and checked by reading, not compiled, not run on hardware
//
#include <hip/hip_runtime.h>
#include <hip/hip_bf16.h>
#include <math.h>

// Spatial2DNATBlock for MI455X (gfx1250, wave32, WMMA).
// All 1x1-conv channel GEMMs go through v_wmma_f32_16x16x32_f16 (f16 in, f32 acc).
// Neighborhood attention (8 heads, hd=8, 8x8 dilated window) stays on VALU.

typedef __attribute__((ext_vector_type(16))) _Float16 v16h;
typedef __attribute__((ext_vector_type(8)))  float    v8f;

#define PIX 16384          // H*W = 128*128 (compile-time: folds e*PIX into load imms)
#define CH  64
#define HDIM 8
#define NHEAD 8

// ---------------------------------------------------------------------------
// LayerNorm2d over channel axis (one thread per pixel). Optional SiLU.
// X, Y are [Cn][P] row-major. In-place safe (each thread owns its column).
// ---------------------------------------------------------------------------
__global__ void ln2d_kernel(const float* __restrict__ X,
                            const float* __restrict__ w,
                            const float* __restrict__ b,
                            float* __restrict__ Y,
                            int Cn, int do_silu)
{
    unsigned p = blockIdx.x * blockDim.x + threadIdx.x;
    if (p >= PIX) return;

    float mean = 0.f;
    for (int c = 0; c < Cn; ++c) mean += X[c * PIX + p];
    mean /= (float)Cn;

    float var = 0.f;
    for (int c = 0; c < Cn; ++c) {
        float d = X[c * PIX + p] - mean;
        var += d * d;
    }
    var /= (float)Cn;
    float inv = rsqrtf(var + 1e-5f);

    for (int c = 0; c < Cn; ++c) {
        float v = (X[c * PIX + p] - mean) * inv * w[c] + b[c];
        if (do_silu) v = v / (1.f + __expf(-v));
        Y[c * PIX + p] = v;
    }
}

// ---------------------------------------------------------------------------
// WMMA GEMM:  D[O][PIX] = Wm[O][KSTEPS*32] * X[KSTEPS*32][PIX] (+ bias[O])
//   MODE 0:  D = acc + bias
//   MODE 1:  D = res + gamma[row] * (acc + bias)   (fused layer-scale residual)
// One wave computes a 16 x (16*NT) strip: A fragment loaded once per K-step
// (as 4x b128) and reused across NT column tiles -> KSTEPS*NT back-to-back
// v_wmma_f32_16x16x32_f16 per wave. All indexing is 32-bit with compile-time
// P so B-row strides (e*PIX*4) land in the 24-bit load immediates (no 64-bit
// address-add chains gating WMMA issue).
// ---------------------------------------------------------------------------
template<int O, int KSTEPS, int MODE, int NT>
__global__ void wmma_gemm_kernel(const float* __restrict__ Wm,
                                 const float* __restrict__ X,
                                 const float* __restrict__ bias,
                                 const float* __restrict__ res,
                                 const float* __restrict__ gamma,
                                 float* __restrict__ D)
{
    constexpr int C    = KSTEPS * 32;
    constexpr int NGRP = PIX / (16 * NT);   // power of two -> shift/mask split
    const unsigned gtid = blockIdx.x * blockDim.x + threadIdx.x;
    const unsigned wave = gtid >> 5;
    const unsigned lane = threadIdx.x & 31;
    const unsigned mt   = wave / NGRP;      // 16-row tile index
    const unsigned ng   = wave % NGRP;      // group of NT 16-col tiles
    if (mt >= (O >> 4)) return;

    const unsigned half = lane >> 4;        // 0: lanes 0-15, 1: lanes 16-31
    const unsigned l16  = lane & 15;

    const unsigned arow = mt * 16 + l16;        // A-matrix row (M)
    const unsigned col0 = ng * (16 * NT) + l16; // first B-matrix col (N)

    v8f acc[NT];
#pragma unroll
    for (int t = 0; t < NT; ++t) acc[t] = (v8f){};

#pragma unroll
    for (int ks = 0; ks < KSTEPS; ++ks) {
        const unsigned k0 = ks * 32;
        // A fragment (16-bit A 16x32 layout, ISA 7.12.2):
        //   lane<16: elems 0-7 -> K=0..7,  elems 8-15 -> K=16..23
        //   lane>=16: shift K by +8
        // Per lane this is two contiguous 8-float runs -> 4x global_load_b128.
        const float* Wr = Wm + arow * C + k0 + half * 8;
        float4 wa0 = *(const float4*)(Wr + 0);
        float4 wa1 = *(const float4*)(Wr + 4);
        float4 wa2 = *(const float4*)(Wr + 16);
        float4 wa3 = *(const float4*)(Wr + 20);
        v16h a;
        a[0]  = (_Float16)wa0.x; a[1]  = (_Float16)wa0.y;
        a[2]  = (_Float16)wa0.z; a[3]  = (_Float16)wa0.w;
        a[4]  = (_Float16)wa1.x; a[5]  = (_Float16)wa1.y;
        a[6]  = (_Float16)wa1.z; a[7]  = (_Float16)wa1.w;
        a[8]  = (_Float16)wa2.x; a[9]  = (_Float16)wa2.y;
        a[10] = (_Float16)wa2.z; a[11] = (_Float16)wa2.w;
        a[12] = (_Float16)wa3.x; a[13] = (_Float16)wa3.y;
        a[14] = (_Float16)wa3.z; a[15] = (_Float16)wa3.w;

#pragma unroll
        for (int t = 0; t < NT; ++t) {
            // B fragment (16-bit B 32x16): elem e -> K = 8*half + e (+8 if e>=8), N = lane%16
            const unsigned boff = (k0 + half * 8) * PIX + col0 + t * 16;
            v16h b;
#pragma unroll
            for (int e = 0; e < 16; ++e)
                b[e] = (_Float16)X[boff + (e + (e >= 8 ? 8 : 0)) * PIX];   // e*PIX*4 -> load imm
            acc[t] = __builtin_amdgcn_wmma_f32_16x16x32_f16(
                /*neg_a=*/false, a, /*neg_b=*/false, b,
                /*c_mod=*/(short)0, acc[t], /*reuse_a=*/false, /*reuse_b=*/false);
        }
    }

    // C/D layout: VGPR r -> M = mt*16 + r + 8*half, N = col0 + t*16
    const unsigned rbase = mt * 16 + half * 8;
    const unsigned obase = rbase * PIX + col0;
#pragma unroll
    for (int r = 0; r < 8; ++r) {
        const unsigned row = rbase + r;
        const float bs = bias[row];
        const float gm = (MODE == 1) ? gamma[row] : 0.f;
#pragma unroll
        for (int t = 0; t < NT; ++t) {
            const unsigned idx = obase + r * PIX + t * 16;  // imm-friendly
            float v = acc[t][r] + bs;
            if (MODE == 1) v = res[idx] + gm * v;
            D[idx] = v;
        }
    }
}

// ---------------------------------------------------------------------------
// Neighborhood attention: 8 heads, hd=8, 8x8 window, stride=2, dilation=2.
// offsets: (i*2 - 8)*2 for i=0..7 -> {-16,-12,-8,-4,0,4,8,12}
// One thread per (head, pixel). q/k pre-LayerNormed.
// ---------------------------------------------------------------------------
__global__ void nat_attn_kernel(const float* __restrict__ q,
                                const float* __restrict__ k,
                                const float* __restrict__ v,
                                float* __restrict__ o)
{
    constexpr int H = 128, W = 128, P = PIX;
    unsigned idx = blockIdx.x * blockDim.x + threadIdx.x;
    if (idx >= NHEAD * P) return;
    const unsigned head = idx >> 14;          // / P
    const unsigned p    = idx & (P - 1);      // % P
    const int py = p >> 7;                    // / W
    const int px = p & (W - 1);               // % W

    const float* qh = q + head * (HDIM * P);
    const float* kh = k + head * (HDIM * P);
    const float* vh = v + head * (HDIM * P);

    float qr[HDIM];
#pragma unroll
    for (int c = 0; c < HDIM; ++c) qr[c] = qh[c * P + p];

    const float scale = 0.35355339059327373f;   // 8^-0.5

    float s[64];
    float m = -INFINITY;
#pragma unroll
    for (int i = 0; i < 8; ++i) {
        int ky  = py + (i * 2 - 8) * 2;
        bool vy = (ky >= 0) && (ky < H);
        int kyc = min(max(ky, 0), H - 1);
#pragma unroll
        for (int j = 0; j < 8; ++j) {
            int kx  = px + (j * 2 - 8) * 2;
            bool vx = (kx >= 0) && (kx < W);
            int kxc = min(max(kx, 0), W - 1);
            unsigned np = (unsigned)(kyc * W + kxc);
            float dot = 0.f;
#pragma unroll
            for (int c = 0; c < HDIM; ++c)
                dot += qr[c] * kh[c * P + np];
            float sc = (vy && vx) ? dot * scale : -INFINITY;
            s[i * 8 + j] = sc;
            m = fmaxf(m, sc);
        }
    }

    float accv[HDIM] = {};
    float sum = 0.f;
#pragma unroll
    for (int i = 0; i < 8; ++i) {
        int ky  = py + (i * 2 - 8) * 2;
        int kyc = min(max(ky, 0), H - 1);
#pragma unroll
        for (int j = 0; j < 8; ++j) {
            int kx  = px + (j * 2 - 8) * 2;
            int kxc = min(max(kx, 0), W - 1);
            unsigned np = (unsigned)(kyc * W + kxc);
            float wgt = __expf(s[i * 8 + j] - m);  // exp(-inf - m) == 0
            sum += wgt;
#pragma unroll
            for (int c = 0; c < HDIM; ++c)
                accv[c] += wgt * vh[c * P + np];
        }
    }

    float inv = 1.f / sum;
#pragma unroll
    for (int c = 0; c < HDIM; ++c)
        o[(head * HDIM + c) * P + p] = accv[c] * inv;
}

// ---------------------------------------------------------------------------
extern "C" void kernel_launch(void* const* d_in, const int* in_sizes, int n_in,
                              void* d_out, int out_size, void* d_ws, size_t ws_size,
                              hipStream_t stream)
{
    const float* x       = (const float*)d_in[0];
    const float* norm1_w = (const float*)d_in[1];
    const float* norm1_b = (const float*)d_in[2];
    const float* qkv_w   = (const float*)d_in[3];
    const float* qkv_b   = (const float*)d_in[4];
    const float* qn_w    = (const float*)d_in[5];
    const float* qn_b    = (const float*)d_in[6];
    const float* kn_w    = (const float*)d_in[7];
    const float* kn_b    = (const float*)d_in[8];
    const float* proj_w  = (const float*)d_in[9];
    const float* proj_b  = (const float*)d_in[10];
    const float* gamma1  = (const float*)d_in[11];
    const float* norm2_w = (const float*)d_in[12];
    const float* norm2_b = (const float*)d_in[13];
    const float* fc1_w   = (const float*)d_in[14];
    const float* fc1_b   = (const float*)d_in[15];
    const float* mn_w    = (const float*)d_in[16];
    const float* mn_b    = (const float*)d_in[17];
    const float* fc2_w   = (const float*)d_in[18];
    const float* fc2_b   = (const float*)d_in[19];
    const float* gamma2  = (const float*)d_in[20];

    const int P = PIX, C = CH;
    const size_t CP = (size_t)C * P;   // 1,048,576 floats

    // Workspace layout (floats), 6*CP = 24 MB total:
    //   [0,   CP) : xn  (later reused as y)
    //   [CP, 4CP) : qkv = q|k|v  (later reused as h1, needs 2CP)
    //   [4CP,5CP) : o
    //   [5CP,6CP) : x1
    float* ws  = (float*)d_ws;
    float* xn  = ws;
    float* qkv = ws + CP;
    float* qp  = qkv;
    float* kp  = qkv + CP;
    float* vp  = qkv + 2 * CP;
    float* o   = ws + 4 * CP;
    float* x1  = ws + 5 * CP;
    float* y   = xn;       // reuse (xn dead after qkv GEMM)
    float* h1  = qkv;      // reuse (qkv dead after attention+proj)
    float* out = (float*)d_out;

    const int LN_BLK = 256;
    dim3 lnGrid((P + LN_BLK - 1) / LN_BLK), lnBlk(LN_BLK);

    constexpr int NT = 4;  // 16-col tiles per wave
    auto gemmGrid = [&](int O) {
        int waves = (O >> 4) * (P / (16 * NT));
        return dim3((waves * 32 + 127) / 128);
    };
    dim3 gBlk(128);

    // 1) xn = LN2d(x)
    ln2d_kernel<<<lnGrid, lnBlk, 0, stream>>>(x, norm1_w, norm1_b, xn, C, 0);
    // 2) qkv = qkv_w * xn + qkv_b      (192x64 @ 64x16384)
    wmma_gemm_kernel<192, 2, 0, NT><<<gemmGrid(3 * C), gBlk, 0, stream>>>(
        qkv_w, xn, qkv_b, nullptr, nullptr, qkv);
    // 3) q = LN2d(q), k = LN2d(k)   (in place)
    ln2d_kernel<<<lnGrid, lnBlk, 0, stream>>>(qp, qn_w, qn_b, qp, C, 0);
    ln2d_kernel<<<lnGrid, lnBlk, 0, stream>>>(kp, kn_w, kn_b, kp, C, 0);
    // 4) neighborhood attention -> o
    {
        int nthr = NHEAD * P;
        nat_attn_kernel<<<dim3((nthr + 255) / 256), dim3(256), 0, stream>>>(
            qp, kp, vp, o);
    }
    // 5) x1 = x + gamma1 * (proj_w * o + proj_b)
    wmma_gemm_kernel<64, 2, 1, NT><<<gemmGrid(C), gBlk, 0, stream>>>(
        proj_w, o, proj_b, x, gamma1, x1);
    // 6) y = LN2d(x1)
    ln2d_kernel<<<lnGrid, lnBlk, 0, stream>>>(x1, norm2_w, norm2_b, y, C, 0);
    // 7) h1 = fc1_w * y + fc1_b       (128x64 @ 64x16384)
    wmma_gemm_kernel<128, 2, 0, NT><<<gemmGrid(2 * C), gBlk, 0, stream>>>(
        fc1_w, y, fc1_b, nullptr, nullptr, h1);
    // 8) h1 = SiLU(LN2d(h1))  over 128 channels, in place
    ln2d_kernel<<<lnGrid, lnBlk, 0, stream>>>(h1, mn_w, mn_b, h1, 2 * C, 1);
    // 9) out = x1 + gamma2 * (fc2_w * h1 + fc2_b)   (64x128 @ 128x16384)
    wmma_gemm_kernel<64, 4, 1, NT><<<gemmGrid(C), gBlk, 0, stream>>>(
        fc2_w, h1, fc2_b, x1, gamma2, out);

    (void)in_sizes; (void)n_in; (void)out_size; (void)ws_size;
}
